// TransformerTower_66838281060981
// MI455X (gfx1250) — matrix operand order repaired and checked by reading, not compiled, not run on hardware
//
#include <hip/hip_runtime.h>
#include <math.h>

// ---------------- problem constants ----------------
#define N_TOK 2048
#define DIMW  768
#define NHEADS 8
#define DQK 128
#define DV  192
#define DP  128
#define MPAIR 128                 // N_TOK / POOL
#define DQKV (NHEADS*DQK + DQK + DV)   // 1344
#define RMS_EPS 1.1920929e-07f
#define QSCALE 0.125f             // 64^-0.5

typedef unsigned int u32;
typedef __bf16 bf16x16 __attribute__((ext_vector_type(16)));
typedef float floatx8 __attribute__((ext_vector_type(8)));
typedef u32   u32x4   __attribute__((ext_vector_type(4)));

union Frag { u32 u[8]; bf16x16 v; };
union AccU { float f[8]; floatx8 v; };

// ---------------- device helpers ----------------
__device__ __forceinline__ u32 packbf2(float lo, float hi) {
    u32 a = __float_as_uint(lo), b = __float_as_uint(hi);
    u32 ra = (a + 0x7FFFu + ((a >> 16) & 1u)) >> 16;
    u32 rb = (b + 0x7FFFu + ((b >> 16) & 1u)) >> 16;
    return (ra & 0xFFFFu) | (rb << 16);
}

// pack 8 consecutive fp32 into 4 bf16-pair dwords
__device__ __forceinline__ void pack8(const float* __restrict__ p, u32* dst) {
    float4 f0 = *(const float4*)p;
    float4 f1 = *(const float4*)(p + 4);
    dst[0] = packbf2(f0.x, f0.y); dst[1] = packbf2(f0.z, f0.w);
    dst[2] = packbf2(f1.x, f1.y); dst[3] = packbf2(f1.z, f1.w);
}

__device__ __forceinline__ floatx8 wmma_bf16(const Frag& a, const Frag& b, floatx8 c) {
    return __builtin_amdgcn_wmma_f32_16x16x32_bf16(false, a.v, false, b.v,
                                                   (short)0, c, false, false);
}

__device__ __forceinline__ float gelu_exact(float v) {
    return 0.5f * v * (1.0f + erff(v * 0.70710678118654752f));
}

// ---------------- generic WMMA GEMM (128x128 tile, double-buffered, pipelined) ----------------
// C[M,N] = act( A[M,K] @ B  + bias ),  B is [K,N] (transB=0) or [N,K] (transB=1)
// 256 threads = 8 waves; each wave computes 32x64 (2x4 16x16 WMMA accumulators).
// K-step 32. LDS tiles stored as bf16 k-pairs with padded stride (bank-conflict free,
// 16B-aligned b128 fragment reads). Global loads for tile k+1 are issued before the
// WMMAs of tile k (double-buffered LDS, one barrier per iteration).
#define GST 20   // dword stride per tile row (16 data + pad)
__launch_bounds__(256)
__global__ void gemm_wmma_kernel(const float* __restrict__ A, long lda,
                                 const float* __restrict__ B, long ldb, int transB,
                                 float* __restrict__ C, long ldc,
                                 int M, int N, int K,
                                 const float* __restrict__ bias, int act)
{
    __shared__ __align__(16) u32 Al[2][128 * GST];
    __shared__ __align__(16) u32 Bl[2][128 * GST];
    const int tid = threadIdx.x;
    const int m0 = blockIdx.y * 128, n0 = blockIdx.x * 128;
    const int wid = tid >> 5, lane = tid & 31;
    const int half = lane >> 4, c = lane & 15;
    const int wm = (wid >> 1) * 32, wn = (wid & 1) * 64;

    AccU acc[2][4];
    for (int mt = 0; mt < 2; ++mt)
        for (int nt = 0; nt < 4; ++nt)
            for (int r = 0; r < 8; ++r) acc[mt][nt].f[r] = 0.f;

    float2 aR[8], bR[8];

    // ---- global tile loaders (into registers) ----
    auto loadA = [&](int kt) {
#pragma unroll
        for (int e = 0; e < 8; ++e) {
            int idx = e * 256 + tid;
            int kk = idx & 15, m = idx >> 4;          // kk fast -> coalesced along k
            int row = m0 + m, k = kt + 2 * kk;
            float2 f = make_float2(0.f, 0.f);
            if (row < M) f = *(const float2*)&A[(long)row * lda + k];
            aR[e] = f;
        }
    };
    auto loadB = [&](int kt) {
        if (!transB) {
#pragma unroll
            for (int e = 0; e < 8; ++e) {
                int idx = e * 256 + tid;
                int n = idx & 127, kk = idx >> 7;     // n fast -> coalesced along n
                int col = n0 + n, k = kt + 2 * kk;
                float2 f = make_float2(0.f, 0.f);
                if (col < N) { f.x = B[(long)k * ldb + col]; f.y = B[(long)(k + 1) * ldb + col]; }
                bR[e] = f;
            }
        } else {
#pragma unroll
            for (int e = 0; e < 8; ++e) {
                int idx = e * 256 + tid;
                int kk = idx & 15, n = idx >> 4;      // kk fast -> coalesced along k row
                int col = n0 + n;
                float2 f = make_float2(0.f, 0.f);
                if (col < N) f = *(const float2*)&B[(long)col * ldb + kt + 2 * kk];
                bR[e] = f;
            }
        }
    };
    auto storeTiles = [&](int p) {
#pragma unroll
        for (int e = 0; e < 8; ++e) {
            int idx = e * 256 + tid;
            Al[p][(idx >> 4) * GST + (idx & 15)] = packbf2(aR[e].x, aR[e].y);
        }
        if (!transB) {
#pragma unroll
            for (int e = 0; e < 8; ++e) {
                int idx = e * 256 + tid;
                Bl[p][(idx & 127) * GST + (idx >> 7)] = packbf2(bR[e].x, bR[e].y);
            }
        } else {
#pragma unroll
            for (int e = 0; e < 8; ++e) {
                int idx = e * 256 + tid;
                Bl[p][(idx >> 4) * GST + (idx & 15)] = packbf2(bR[e].x, bR[e].y);
            }
        }
    };

    // ---- pipelined main loop ----
    loadA(0); loadB(0);
    int p = 0;
    for (int kt = 0; kt < K; kt += 32) {
        storeTiles(p);
        __syncthreads();
        if (kt + 32 < K) { loadA(kt + 32); loadB(kt + 32); }   // overlap with WMMAs below
        // fragments: A vgpr i -> k = (i<4?0:16)+half*8+2(i&3); B vgpr j -> k = (j<4?0:16)+half*8+2(j&3)
        Frag a[2], b[4];
#pragma unroll
        for (int mt = 0; mt < 2; ++mt) {
            const u32* ap = &Al[p][(wm + mt * 16 + c) * GST + half * 4];
            *(u32x4*)&a[mt].u[0] = *(const u32x4*)ap;
            *(u32x4*)&a[mt].u[4] = *(const u32x4*)(ap + 8);
        }
#pragma unroll
        for (int nt = 0; nt < 4; ++nt) {
            const u32* bp = &Bl[p][(wn + nt * 16 + c) * GST + half * 4];
            *(u32x4*)&b[nt].u[0] = *(const u32x4*)bp;
            *(u32x4*)&b[nt].u[4] = *(const u32x4*)(bp + 8);
        }
#pragma unroll
        for (int mt = 0; mt < 2; ++mt)
#pragma unroll
            for (int nt = 0; nt < 4; ++nt)
                acc[mt][nt].v = wmma_bf16(a[mt], b[nt], acc[mt][nt].v);
        p ^= 1;
    }
    // ---- epilogue: C layout vgpr r -> m = r + 8*half, n = lane&15 ----
    for (int mt = 0; mt < 2; ++mt)
        for (int nt = 0; nt < 4; ++nt)
            for (int r = 0; r < 8; ++r) {
                int m = m0 + wm + mt * 16 + r + half * 8;
                int n = n0 + wn + nt * 16 + c;
                if (m < M && n < N) {
                    float v = acc[mt][nt].f[r];
                    if (bias) v += bias[n];
                    if (act == 1) v = fmaxf(v, 0.f);
                    C[(long)m * ldc + n] = v;
                }
            }
}

// ---------------- fused pairwise row-attention ----------------
// one block per 128-row slice; 8 waves; sim + softmax (in regs) + probs@V + residual
#define PAST 68   // dword stride for probs staging (64 data + pad)
__launch_bounds__(256)
__global__ void pair_attn_kernel(const float* __restrict__ pqk,   // [16384,256]: q|k
                                 const float* __restrict__ pv,    // [16384,128]
                                 float* __restrict__ pairwise)    // [16384,128] in/out
{
    __shared__ __align__(16) u32 pA[128 * PAST];   // probs staged as bf16 A-fragments
    const long base = (long)blockIdx.x * 128;
    const int tid = threadIdx.x, wid = tid >> 5, lane = tid & 31;
    const int half = lane >> 4, c = lane & 15;
    const int mrow = wid * 16;

    AccU acc[8];
    for (int nt = 0; nt < 8; ++nt)
        for (int r = 0; r < 8; ++r) acc[nt].f[r] = 0.f;

    // sim = q @ k^T  (K=128, 4 steps of 32)
    const float* qrow = pqk + (base + mrow + c) * 256;
    for (int s = 0; s < 4; ++s) {
        Frag a;
        pack8(qrow + s * 32 + half * 8, &a.u[0]);
        pack8(qrow + s * 32 + 16 + half * 8, &a.u[4]);
#pragma unroll
        for (int nt = 0; nt < 8; ++nt) {
            Frag b;
            const float* kp = pqk + (base + nt * 16 + c) * 256 + 128 + s * 32 + half * 8;
            pack8(kp, &b.u[0]); pack8(kp + 16, &b.u[4]);
            acc[nt].v = wmma_bf16(a, b, acc[nt].v);
        }
    }
    // row softmax fully in registers (row lives in 16 lanes of one half)
    for (int r = 0; r < 8; ++r) {
        float mx = -1e30f;
        for (int nt = 0; nt < 8; ++nt) mx = fmaxf(mx, acc[nt].f[r]);
        for (int o = 1; o < 16; o <<= 1) mx = fmaxf(mx, __shfl_xor(mx, o, 32));
        float sum = 0.f;
        for (int nt = 0; nt < 8; ++nt) { float p = __expf(acc[nt].f[r] - mx); acc[nt].f[r] = p; sum += p; }
        for (int o = 1; o < 16; o <<= 1) sum += __shfl_xor(sum, o, 32);
        float inv = 1.f / sum;
        for (int nt = 0; nt < 8; ++nt) acc[nt].f[r] *= inv;
    }
    // restage probs (C layout) -> bf16 A layout in LDS (wave-private rows)
    for (int r = 0; r < 8; ++r) {
        int m = mrow + r + half * 8;
        for (int nt = 0; nt < 8; ++nt) {
            float p = acc[nt].f[r];
            float pn = __shfl_xor(p, 1, 32);
            if (!(c & 1)) pA[m * PAST + ((nt * 16 + c) >> 1)] = packbf2(p, pn);
        }
    }
    // out = probs @ v  + residual
    AccU acc2[8];
    for (int nt = 0; nt < 8; ++nt)
        for (int r = 0; r < 8; ++r) acc2[nt].f[r] = 0.f;
    for (int s = 0; s < 4; ++s) {
        Frag a;
        const u32* ap = &pA[(mrow + c) * PAST + s * 16 + half * 4];
        *(u32x4*)&a.u[0] = *(const u32x4*)ap;
        *(u32x4*)&a.u[4] = *(const u32x4*)(ap + 8);
#pragma unroll
        for (int nt = 0; nt < 8; ++nt) {
            Frag b;
            const float* vp = pv + (base + nt * 16 + c) * 128 + s * 32 + half * 8;
            pack8(vp, &b.u[0]); pack8(vp + 16, &b.u[4]);
            acc2[nt].v = wmma_bf16(a, b, acc2[nt].v);
        }
    }
    for (int nt = 0; nt < 8; ++nt)
        for (int r = 0; r < 8; ++r) {
            int m = mrow + r + half * 8, n = nt * 16 + c;
            long gi = (base + m) * 128 + n;
            pairwise[gi] = acc2[nt].f[r] + pairwise[gi];
        }
}

// ---------------- RMSNorm (fused weight/scale/gelu/residual) ----------------
__launch_bounds__(256)
__global__ void rmsnorm_kernel(const float* __restrict__ in, long istride,
                               float* __restrict__ out, long ostride,
                               const float* __restrict__ w, int dim,
                               float scale, int do_gelu,
                               const float* __restrict__ residual)
{
    const long row = blockIdx.x;
    const float* ip = in + row * istride;
    float ss = 0.f;
    for (int d = threadIdx.x; d < dim; d += 256) { float v = ip[d]; ss += v * v; }
    for (int o = 16; o; o >>= 1) ss += __shfl_xor(ss, o, 32);
    __shared__ float red[8];
    __shared__ float stot;
    if ((threadIdx.x & 31) == 0) red[threadIdx.x >> 5] = ss;
    __syncthreads();
    if (threadIdx.x == 0) { float t = 0; for (int i = 0; i < 8; ++i) t += red[i]; stot = t; }
    __syncthreads();
    float rinv = rsqrtf(stot / (float)dim + RMS_EPS);
    for (int d = threadIdx.x; d < dim; d += 256) {
        float v = ip[d] * rinv * w[d] * scale;
        if (do_gelu) v = gelu_exact(v);
        out[row * ostride + d] = residual ? residual[row * ostride + d] + v : v;
    }
}

// ---------------- softclamp + pooled bias + softmax (one head) ----------------
__launch_bounds__(256)
__global__ void softclamp_softmax_kernel(float* __restrict__ sim,
                                         const float* __restrict__ biasb, // [16384,8]
                                         int head)
{
    const int i = blockIdx.x;
    const int tid = threadIdx.x;
    const float* bb = biasb + (long)(i >> 4) * 128 * 8 + head;
    float vals[8];
    float mx = -1e30f;
    for (int e = 0; e < 8; ++e) {
        int j = e * 256 + tid;
        float s = sim[(long)i * 2048 + j] + bb[(j >> 4) * 8];
        float t = tanhf(s * 0.2f) * 5.0f;
        vals[e] = t;
        mx = fmaxf(mx, t);
    }
    __shared__ float red[8], red2[8];
    for (int o = 16; o; o >>= 1) mx = fmaxf(mx, __shfl_xor(mx, o, 32));
    if ((tid & 31) == 0) red[tid >> 5] = mx;
    __syncthreads();
    mx = red[0];
    for (int w = 1; w < 8; ++w) mx = fmaxf(mx, red[w]);
    float sum = 0.f;
    for (int e = 0; e < 8; ++e) { vals[e] = __expf(vals[e] - mx); sum += vals[e]; }
    for (int o = 16; o; o >>= 1) sum += __shfl_xor(sum, o, 32);
    if ((tid & 31) == 0) red2[tid >> 5] = sum;
    __syncthreads();
    sum = 0.f;
    for (int w = 0; w < 8; ++w) sum += red2[w];
    float inv = 1.f / sum;
    for (int e = 0; e < 8; ++e) sim[(long)i * 2048 + e * 256 + tid] = vals[e] * inv;
}

// ---------------- small elementwise kernels ----------------
__global__ void rope_kernel(float* __restrict__ buf, long rows, int posmask)
{
    long idx = (long)blockIdx.x * 256 + threadIdx.x;
    if (idx >= rows * 64) return;
    long row = idx >> 6; int j = idx & 63;
    float n = (float)(row & posmask);
    float e = 1.0f + (float)j * (8128.0f / 63.0f);          // linspace(1, 8129, 64)
    float p = __expf(e * 2.302585092994046f);               // 10^e (inf for big e -> invf=0)
    float invf = 1.0f / ((float)j + p);
    float a = n * invf;
    float cc = cosf(a), ssn = sinf(a);
    float* r = buf + row * 128;
    float t0 = r[j], t1 = r[j + 64];
    r[j]      = t0 * cc - t1 * ssn;
    r[j + 64] = t1 * cc + t0 * ssn;
}

__global__ void pool_mean_kernel(const float* __restrict__ x, float* __restrict__ pooled)
{
    int idx = blockIdx.x * 256 + threadIdx.x;
    if (idx >= MPAIR * DIMW) return;
    int mi = idx / DIMW, d = idx - mi * DIMW;
    float s = 0.f;
    for (int p = 0; p < 16; ++p) s += x[(long)(mi * 16 + p) * DIMW + d];
    pooled[idx] = s * 0.0625f;
}

__global__ void gelu_kernel(const float* __restrict__ in, float* __restrict__ out, long n)
{
    long i = (long)blockIdx.x * 256 + threadIdx.x;
    if (i < n) out[i] = gelu_exact(in[i]);
}

__global__ void outer_sum_kernel(const float* __restrict__ oqk,  // [128,256] = oq|ok
                                 float* __restrict__ pw, int addOld)
{
    long idx = (long)blockIdx.x * 256 + threadIdx.x;
    if (idx >= (long)MPAIR * MPAIR * DP) return;
    int d = idx & 127;
    long ij = idx >> 7;
    int j = ij & 127, i = (int)(ij >> 7);
    float v = oqk[i * 256 + d] + oqk[j * 256 + 128 + d];
    pw[idx] = addOld ? pw[idx] + v : v;
}

// ---------------- host-side launch helpers ----------------
static void gemm(hipStream_t s, const float* A, long lda, const float* B, long ldb, int tB,
                 float* C, long ldc, int M, int N, int K, const float* bias, int act)
{
    dim3 g((N + 127) / 128, (M + 127) / 128);
    gemm_wmma_kernel<<<g, dim3(256), 0, s>>>(A, lda, B, ldb, tB, C, ldc, M, N, K, bias, act);
}

static void rms(hipStream_t s, const float* in, long istride, float* out, long ostride,
                const float* w, int dim, long rows, float scale, int do_gelu, const float* resid)
{
    rmsnorm_kernel<<<dim3((unsigned)rows), dim3(256), 0, s>>>(in, istride, out, ostride,
                                                              w, dim, scale, do_gelu, resid);
}

// ---------------- orchestration ----------------
extern "C" void kernel_launch(void* const* d_in, const int* in_sizes, int n_in,
                              void* d_out, int out_size, void* d_ws, size_t ws_size,
                              hipStream_t stream)
{
    (void)in_sizes; (void)n_in; (void)out_size; (void)ws_size;
    const float* single_in   = (const float*)d_in[0];
    const float* a_pre       = (const float*)d_in[1];
    const float* a_post      = (const float*)d_in[2];
    const float* qkv_w       = (const float*)d_in[3];
    const float* qn_w        = (const float*)d_in[4];
    const float* kn_w        = (const float*)d_in[5];
    const float* vn_w        = (const float*)d_in[6];
    const float* out_w       = (const float*)d_in[7];
    const float* bias_norm_w = (const float*)d_in[8];
    const float* bias_w      = (const float*)d_in[9];
    const float* f_pre       = (const float*)d_in[10];
    const float* f_post      = (const float*)d_in[11];
    const float* ff_w1       = (const float*)d_in[12];
    const float* ff_b1       = (const float*)d_in[13];
    const float* ff_w2       = (const float*)d_in[14];
    const float* ff_b2       = (const float*)d_in[15];
    const float* sp_pre      = (const float*)d_in[16];
    const float* sp_outer    = (const float*)d_in[17];
    const float* pa_pre      = (const float*)d_in[18];
    const float* pa_qk       = (const float*)d_in[19];
    const float* pa_v        = (const float*)d_in[20];
    const float* pa_v_b      = (const float*)d_in[21];
    const float* pf_pre      = (const float*)d_in[22];
    const float* pf_w1       = (const float*)d_in[23];
    const float* pf_b1       = (const float*)d_in[24];
    const float* pf_w2       = (const float*)d_in[25];
    const float* pf_b2       = (const float*)d_in[26];

    float* single = (float*)d_out;   // [2048,768], updated in place
    hipMemcpyAsync(single, single_in, (size_t)N_TOK * DIMW * sizeof(float),
                   hipMemcpyDeviceToDevice, stream);

    float* ws = (float*)d_ws;
    size_t off = 0;
    auto alloc = [&](size_t n) { float* p = ws + off; off += n; return p; };
    float* xb      = alloc((size_t)N_TOK * DIMW);
    float* hbuf    = alloc((size_t)N_TOK * 2 * DIMW);
    float* qkvb    = alloc((size_t)N_TOK * DQKV);
    float* qbuf    = alloc((size_t)NHEADS * N_TOK * DQK);
    float* kbuf    = alloc((size_t)N_TOK * DQK);
    float* vbuf    = alloc((size_t)N_TOK * DV);
    float* sim     = alloc((size_t)N_TOK * N_TOK);
    float* obuf    = alloc((size_t)N_TOK * NHEADS * DV);
    float* t768    = alloc((size_t)N_TOK * DIMW);
    float* pairwise= alloc((size_t)MPAIR * MPAIR * DP);
    float* pairx   = alloc((size_t)MPAIR * MPAIR * DP);
    float* pqk     = alloc((size_t)MPAIR * MPAIR * 2 * DP);
    float* pv      = alloc((size_t)MPAIR * MPAIR * DP);
    float* pooled  = alloc((size_t)MPAIR * DIMW);
    float* pooledg = alloc((size_t)MPAIR * DIMW);
    float* oqk     = alloc((size_t)MPAIR * 2 * DP);
    float* biasb   = alloc((size_t)MPAIR * MPAIR * NHEADS);

    const long NP = (long)MPAIR * MPAIR;    // 16384 pairwise rows

    for (int l = 0; l < 4; ++l) {
        if ((l & 1) == 0) {
            int pi = l / 2;
            // SingleToPairwise
            rms(stream, single, DIMW, xb, DIMW, sp_pre + (size_t)pi * DIMW, DIMW, N_TOK, 1.f, 0, nullptr);
            pool_mean_kernel<<<(MPAIR * DIMW + 255) / 256, 256, 0, stream>>>(xb, pooled);
            gelu_kernel<<<(MPAIR * DIMW + 255) / 256, 256, 0, stream>>>(pooled, pooledg, (long)MPAIR * DIMW);
            gemm(stream, pooledg, DIMW, sp_outer + (size_t)pi * DIMW * 256, 256, 0,
                 oqk, 256, MPAIR, 256, DIMW, nullptr, 0);
            outer_sum_kernel<<<(unsigned)((NP * DP + 255) / 256), 256, 0, stream>>>(oqk, pairwise, pi > 0 ? 1 : 0);
            // PairwiseRowAttention
            rms(stream, pairwise, DP, pairx, DP, pa_pre + (size_t)pi * DP, DP, NP, 1.f, 0, nullptr);
            gemm(stream, pairx, DP, pa_qk + (size_t)pi * DP * 256, 256, 0,
                 pqk, 256, (int)NP, 256, DP, nullptr, 0);
            gemm(stream, pairx, DP, pa_v + (size_t)pi * DP * DP, DP, 0,
                 pv, DP, (int)NP, DP, DP, pa_v_b + (size_t)pi * DP, 0);
            pair_attn_kernel<<<MPAIR, 256, 0, stream>>>(pqk, pv, pairwise);
            // pairwise FeedForward (no residual)
            rms(stream, pairwise, DP, pairx, DP, pf_pre + (size_t)pi * DP, DP, NP, 1.f, 0, nullptr);
            gemm(stream, pairx, DP, pf_w1 + (size_t)pi * DP * 256, 256, 0,
                 pqk, 256, (int)NP, 256, DP, pf_b1 + (size_t)pi * 256, 1);
            gemm(stream, pqk, 256, pf_w2 + (size_t)pi * 256 * DP, DP, 0,
                 pairwise, DP, (int)NP, DP, 256, pf_b2 + (size_t)pi * DP, 0);
        }
        // ---- attention block ----
        rms(stream, single, DIMW, xb, DIMW, a_pre + (size_t)l * DIMW, DIMW, N_TOK, 1.f, 0, nullptr);
        gemm(stream, xb, DIMW, qkv_w + (size_t)l * DIMW * DQKV, DQKV, 0,
             qkvb, DQKV, N_TOK, DQKV, DIMW, nullptr, 0);
        for (int h = 0; h < NHEADS; ++h)
            rms(stream, qkvb + h * DQK, DQKV, qbuf + (size_t)h * N_TOK * DQK, DQK,
                qn_w + (size_t)l * DQK, DQK, N_TOK, QSCALE, 0, nullptr);
        rms(stream, qkvb + NHEADS * DQK, DQKV, kbuf, DQK, kn_w + (size_t)l * DQK, DQK, N_TOK, 1.f, 0, nullptr);
        rms(stream, qkvb + NHEADS * DQK + DQK, DQKV, vbuf, DV, vn_w + (size_t)l * DV, DV, N_TOK, 1.f, 0, nullptr);
        rope_kernel<<<(unsigned)(((long)NHEADS * N_TOK * 64 + 255) / 256), 256, 0, stream>>>(
            qbuf, (long)NHEADS * N_TOK, N_TOK - 1);
        rope_kernel<<<(unsigned)(((long)N_TOK * 64 + 255) / 256), 256, 0, stream>>>(
            kbuf, (long)N_TOK, N_TOK - 1);
        // attention bias from pairwise
        rms(stream, pairwise, DP, pairx, DP, bias_norm_w + (size_t)l * DP, DP, NP, 1.f, 1, nullptr);
        gemm(stream, pairx, DP, bias_w + (size_t)l * DP * NHEADS, NHEADS, 0,
             biasb, NHEADS, (int)NP, NHEADS, DP, nullptr, 0);
        // per-head: sim = q k^T ; softclamp+bias+softmax ; o = attn v
        for (int h = 0; h < NHEADS; ++h) {
            gemm(stream, qbuf + (size_t)h * N_TOK * DQK, DQK, kbuf, DQK, 1,
                 sim, N_TOK, N_TOK, N_TOK, DQK, nullptr, 0);
            softclamp_softmax_kernel<<<N_TOK, 256, 0, stream>>>(sim, biasb, h);
            gemm(stream, sim, N_TOK, vbuf, DV, 0,
                 obuf + h * DV, NHEADS * DV, N_TOK, DV, N_TOK, nullptr, 0);
        }
        gemm(stream, obuf, NHEADS * DV, out_w + (size_t)l * NHEADS * DV * DIMW, DIMW, 0,
             t768, DIMW, N_TOK, DIMW, NHEADS * DV, nullptr, 0);
        rms(stream, t768, DIMW, single, DIMW, a_post + (size_t)l * DIMW, DIMW, N_TOK, 1.f, 0, single);
        // ---- feedforward block ----
        rms(stream, single, DIMW, xb, DIMW, f_pre + (size_t)l * DIMW, DIMW, N_TOK, 1.f, 0, nullptr);
        gemm(stream, xb, DIMW, ff_w1 + (size_t)l * DIMW * 2 * DIMW, 2 * DIMW, 0,
             hbuf, 2 * DIMW, N_TOK, 2 * DIMW, DIMW, ff_b1 + (size_t)l * 2 * DIMW, 1);
        gemm(stream, hbuf, 2 * DIMW, ff_w2 + (size_t)l * 2 * DIMW * DIMW, DIMW, 0,
             t768, DIMW, N_TOK, DIMW, 2 * DIMW, ff_b2 + (size_t)l * DIMW, 0);
        rms(stream, t768, DIMW, single, DIMW, f_post + (size_t)l * DIMW, DIMW, N_TOK, 1.f, 0, single);
    }
}
